// MicroEvent_67705864454749
// MI455X (gfx1250) — matrix operand order repaired and checked by reading, not compiled, not run on hardware
//
#include <hip/hip_runtime.h>
#include <hip/hip_bf16.h>
#include <math.h>

// ---------------------------------------------------------------------------
// MicroEvent pipeline for MI455X (gfx1250, wave32, WMMA, async-LDS copies).
//  - GEMMs: bf16 V_WMMA_F32_16X16X32_BF16; operands pre-converted to bf16
//    (halves weight bandwidth) and B pre-transposed to [N][K] so every LDS
//    stage is a layout-preserving copy -> GLOBAL_LOAD_ASYNC_TO_LDS_B128,
//    double-buffered with s_wait_asynccnt.
//  - 65536-pt real FFTs: packed 32768-pt complex FFT, fully LDS-resident
//    (256KB of the WGP's 320KB LDS); rows stream in/out via async LDS copies.
// ---------------------------------------------------------------------------

typedef __attribute__((ext_vector_type(16))) __bf16 v16bf;
typedef __attribute__((ext_vector_type(8)))  float  v8f;
typedef __attribute__((ext_vector_type(4)))  int    v4i;

union Frag16 { v16bf v; unsigned u[8]; };
union Acc8   { v8f   v; float    f[8]; };

static constexpr int kBE   = 512;     // B*E rows
static constexpr int kL    = 256;     // latent / channel dim
static constexpr int kRES  = 32768;
static constexpr int kIMP  = 8192;
static constexpr int kNB   = 64;      // resonance bank rows
static constexpr int kMF   = 32768;   // packed complex length for 65536-pt rFFT
static constexpr int kNF   = 65536;

// ------------------------- async LDS copy plumbing -------------------------
#if defined(__has_builtin)
#if __has_builtin(__builtin_amdgcn_global_load_async_to_lds_b128)
#define HAVE_ASYNC_LD 1
#else
#define HAVE_ASYNC_LD 0
#endif
#if __has_builtin(__builtin_amdgcn_global_store_async_from_lds_b128)
#define HAVE_ASYNC_ST 1
#else
#define HAVE_ASYNC_ST 0
#endif
#else
#define HAVE_ASYNC_LD 0
#define HAVE_ASYNC_ST 0
#endif

// builtin signature (from hipcc diagnostics): (v4i AS1*, v4i AS3*, imm, imm)
typedef __attribute__((address_space(1))) v4i* gv4i_t;
typedef __attribute__((address_space(3))) v4i* lv4i_t;

__device__ __forceinline__ void async_copy16(const void* g, void* l) {
#if HAVE_ASYNC_LD
  __builtin_amdgcn_global_load_async_to_lds_b128((gv4i_t)g, (lv4i_t)l, 0, 0);
#else
  *(uint4*)l = *(const uint4*)g;
#endif
}

__device__ __forceinline__ void async_store16(void* g, void* l) {
#if HAVE_ASYNC_ST
  __builtin_amdgcn_global_store_async_from_lds_b128((gv4i_t)g, (lv4i_t)l, 0, 0);
#else
  *(uint4*)g = *(const uint4*)l;
#endif
}

template <int N> __device__ __forceinline__ void async_wait() {
#if HAVE_ASYNC_LD || HAVE_ASYNC_ST
#if __has_builtin(__builtin_amdgcn_s_wait_asynccnt)
  __builtin_amdgcn_s_wait_asynccnt(N);
#else
  asm volatile("s_wait_asynccnt %0" ::"i"(N) : "memory");
#endif
#endif
}

// ---------------------------------------------------------------------------
__device__ __forceinline__ unsigned short f2bf(float x) {
  unsigned u = __float_as_uint(x);
  u += 0x7FFFu + ((u >> 16) & 1u);          // round-to-nearest-even
  return (unsigned short)(u >> 16);
}

__device__ __forceinline__ float2 cmul(float2 a, float2 b) {
  return make_float2(a.x * b.x - a.y * b.y, a.x * b.y + a.y * b.x);
}

// ===========================================================================
// bf16 convert (flat) and convert+transpose ([K][N] f32 -> [N][K] bf16).
// ===========================================================================
__global__ __launch_bounds__(256)
void k_cvt(const float* __restrict__ src, unsigned short* __restrict__ dst,
           size_t n) {
  size_t i = (size_t)blockIdx.x * 256 + threadIdx.x;
  if (i < n) dst[i] = f2bf(src[i]);
}

__global__ __launch_bounds__(256)
void k_cvt_t(const float* __restrict__ src, unsigned short* __restrict__ dst,
             int K, int N) {
  __shared__ float tile[32][33];
  const int nb = blockIdx.x * 32, kb = blockIdx.y * 32;
  const int tx = threadIdx.x & 31, ty = threadIdx.x >> 5;
  for (int j = ty; j < 32; j += 8)
    tile[j][tx] = src[(size_t)(kb + j) * N + nb + tx];
  __syncthreads();
  for (int j = ty; j < 32; j += 8)
    dst[(size_t)(nb + j) * K + kb + tx] = f2bf(tile[tx][j]);
}

// ===========================================================================
// bf16 GEMM, C[M,N] = A[M,K] * Bt[N,K]^T.  Block 256 thr (8 waves),
// tile 64(M) x 128(N), K-step 32, double-buffered async global->LDS staging.
// ===========================================================================
#define GT_M 64
#define GT_N 128
#define GT_K 32

__global__ __launch_bounds__(256)
void k_wmma_gemm_bf16(const unsigned short* __restrict__ A,   // [M,K] bf16
                      const unsigned short* __restrict__ Bt,  // [N,K] bf16
                      float* __restrict__ C, int M, int N, int K) {
  __shared__ __align__(16) unsigned short As[2][GT_M][GT_K];
  __shared__ __align__(16) unsigned short Bs[2][GT_N][GT_K];
  const int tid  = threadIdx.x;
  const int lane = tid & 31, w = tid >> 5;
  const int wm = w & 3, wn = w >> 2;
  const int bm = blockIdx.y * GT_M, bn = blockIdx.x * GT_N;

  Acc8 acc[4];
#pragma unroll
  for (int t = 0; t < 4; ++t)
#pragma unroll
    for (int r = 0; r < 8; ++r) acc[t].f[r] = 0.f;

  const int arow = tid >> 2, apart = (tid & 3) * 8;     // one 16B chunk / thread
  auto issue = [&](int buf, int k0) {
    async_copy16(A + (size_t)(bm + arow) * K + k0 + apart, &As[buf][arow][apart]);
#pragma unroll
    for (int c = 0; c < 2; ++c) {                        // two 16B chunks for B
      int idx  = tid + c * 256;
      int brow = idx >> 2, bpart = (idx & 3) * 8;
      async_copy16(Bt + (size_t)(bn + brow) * K + k0 + bpart, &Bs[buf][brow][bpart]);
    }
  };

  const int nt = K / GT_K;
  issue(0, 0);
  for (int t = 0; t < nt; ++t) {
    const int cur = t & 1;
    if (t + 1 < nt) { issue(cur ^ 1, (t + 1) * GT_K); async_wait<3>(); }
    else            { async_wait<0>(); }
    __syncthreads();

    // A fragment (ISA 7.12.2, 16-bit A 16x32): lane l holds row (l&15),
    // K-half (l>>4): vgpr v -> K = (v>=4?16:0) + 8*(l>>4) + 2*(v&3)
    Frag16 af;
    {
      const int row = wm * 16 + (lane & 15);
      const int kh  = (lane >> 4) * 8;
#pragma unroll
      for (int v = 0; v < 8; ++v) {
        int kk = ((v >= 4) ? 16 : 0) + kh + ((v & 3) << 1);
        af.u[v] = *(const unsigned*)&As[cur][row][kk];
      }
    }
#pragma unroll
    for (int t4 = 0; t4 < 4; ++t4) {
      // B fragment (32x16): lane l covers N=(l&15), same K order as A
      Frag16 bfv;
      const int ncol = wn * 64 + t4 * 16 + (lane & 15);
      const int kb   = (lane >> 4) * 8;
#pragma unroll
      for (int v = 0; v < 8; ++v)
        bfv.u[v] = *(const unsigned*)&Bs[cur][ncol][((v >= 4) ? 16 : 0) + kb + ((v & 3) << 1)];
      acc[t4].v = __builtin_amdgcn_wmma_f32_16x16x32_bf16(
          false, af.v, false, bfv.v, (short)0, acc[t4].v, false, false);
    }
    __syncthreads();
  }

  // C/D layout: vgpr r -> M = r + 8*(lane>=16); N = lane&15
  const int nloc = lane & 15, mhi = lane >> 4;
#pragma unroll
  for (int t = 0; t < 4; ++t) {
    int col = bn + wn * 64 + t * 16 + nloc;
#pragma unroll
    for (int r = 0; r < 8; ++r) {
      int row = bm + wm * 16 + mhi * 8 + r;
      C[(size_t)row * N + col] = acc[t].f[r];
    }
  }
}

// ===========================================================================
// Fused LinearOutputStack: in-proj, 3x (GEMM -> LN -> leaky), out-proj.
// One block owns 16 rows end-to-end; activations resident in LDS (bf16+f32).
// mode 0: ReLU epilogue (sel2, out_dim=64).  mode 1: softmax (mix, out_dim=3).
// ===========================================================================
__device__ void stack_gemm_tile(const unsigned short Hb[][kL],
                                unsigned short Bs[][32],
                                const float* __restrict__ W,
                                const float* __restrict__ bias,
                                float Hf[][kL]) {
  const int tid = threadIdx.x, lane = tid & 31, w = tid >> 5;
  Acc8 acc[2];
#pragma unroll
  for (int t = 0; t < 2; ++t)
#pragma unroll
    for (int r = 0; r < 8; ++r) acc[t].f[r] = 0.f;

  for (int k0 = 0; k0 < kL; k0 += 32) {
    for (int i = tid; i < 32 * kL; i += 256) {      // stage W[k0:+32, :] -> Bs[n][k]
      int kr = i >> 8, nc = i & 255;
      Bs[nc][kr] = f2bf(W[(size_t)(k0 + kr) * kL + nc]);
    }
    __syncthreads();
    Frag16 af;
    {
      const int row = lane & 15;
      const int kh  = (lane >> 4) * 8;
#pragma unroll
      for (int v = 0; v < 8; ++v) {
        int kk = k0 + ((v >= 4) ? 16 : 0) + kh + ((v & 3) << 1);
        af.u[v] = *(const unsigned*)&Hb[row][kk];
      }
    }
#pragma unroll
    for (int t = 0; t < 2; ++t) {
      Frag16 bfv;
      const int ncol = w * 32 + t * 16 + (lane & 15);
      const int kb   = (lane >> 4) * 8;
#pragma unroll
      for (int v = 0; v < 8; ++v)
        bfv.u[v] = *(const unsigned*)&Bs[ncol][((v >= 4) ? 16 : 0) + kb + ((v & 3) << 1)];
      acc[t].v = __builtin_amdgcn_wmma_f32_16x16x32_bf16(
          false, af.v, false, bfv.v, (short)0, acc[t].v, false, false);
    }
    __syncthreads();
  }
  const int nloc = lane & 15, mhi = lane >> 4;
#pragma unroll
  for (int t = 0; t < 2; ++t) {
    int col = w * 32 + t * 16 + nloc;
#pragma unroll
    for (int r = 0; r < 8; ++r)
      Hf[mhi * 8 + r][col] = acc[t].f[r] + bias[col];
  }
  __syncthreads();
}

__global__ __launch_bounds__(256)
void k_stack(const float* __restrict__ X,
             const float* __restrict__ Win,  const float* __restrict__ b_in,
             const float* __restrict__ Wh,   const float* __restrict__ bh,
             const float* __restrict__ g,    const float* __restrict__ beta,
             const float* __restrict__ Wout, const float* __restrict__ bout,
             float* __restrict__ Out, int out_dim, int mode) {
  __shared__ unsigned short Hb[16][kL];   // bf16 activations (WMMA A operand)
  __shared__ unsigned short Bs[kL][32];   // bf16 weight tile, N-major
  __shared__ float          Hf[16][kL];   // f32 activations
  __shared__ float2         red2[16][16];
  __shared__ float2         stat[16];
  const int tid = threadIdx.x;
  const int bm  = blockIdx.x * 16;

  for (int i = tid; i < 16 * kL; i += 256) {
    int r = i >> 8, c = i & 255;
    Hb[r][c] = f2bf(X[(size_t)(bm + r) * kL + c]);
  }
  __syncthreads();

  // in-projection (no LN / no activation)
  stack_gemm_tile(Hb, Bs, Win, b_in, Hf);
  for (int i = tid; i < 16 * kL; i += 256) {
    int r = i >> 8, c = i & 255;
    Hb[r][c] = f2bf(Hf[r][c]);
  }
  __syncthreads();

  for (int l = 0; l < 3; ++l) {
    stack_gemm_tile(Hb, Bs, Wh + (size_t)l * kL * kL, bh + l * kL, Hf);
    {   // LayerNorm over 256 channels: 16 threads per row
      const int row = tid >> 4, idx = tid & 15;
      float s = 0.f, q = 0.f;
#pragma unroll
      for (int j = 0; j < 16; ++j) {
        float v = Hf[row][idx * 16 + j];
        s += v; q += v * v;
      }
      red2[row][idx] = make_float2(s, q);
      __syncthreads();
      if (idx == 0) {
        float S = 0.f, Q = 0.f;
#pragma unroll
        for (int j = 0; j < 16; ++j) { S += red2[row][j].x; Q += red2[row][j].y; }
        float mu  = S * (1.0f / 256.0f);
        float var = Q * (1.0f / 256.0f) - mu * mu;
        stat[row] = make_float2(mu, rsqrtf(var + 1e-5f));
      }
      __syncthreads();
      const float* gl = g + l * kL;
      const float* bl = beta + l * kL;
      for (int i = tid; i < 16 * kL; i += 256) {
        int r = i >> 8, c = i & 255;
        float z = (Hf[r][c] - stat[r].x) * stat[r].y * gl[c] + bl[c];
        z = (z > 0.f) ? z : 0.2f * z;                 // leaky ReLU
        Hf[r][c] = z;
        Hb[r][c] = f2bf(z);
      }
      __syncthreads();
    }
  }

  if (mode == 0) {
    for (int i = tid; i < 16 * out_dim; i += 256) {
      int r = i / out_dim, od = i % out_dim;
      float a = bout[od];
      for (int c = 0; c < kL; ++c) a += Hf[r][c] * Wout[(size_t)c * out_dim + od];
      Out[(size_t)(bm + r) * out_dim + od] = fmaxf(a, 0.f);
    }
  } else {
    if (tid < 16) {
      float o[3];
#pragma unroll
      for (int od = 0; od < 3; ++od) {
        float a = bout[od];
        for (int c = 0; c < kL; ++c) a += Hf[tid][c] * Wout[(size_t)c * 3 + od];
        o[od] = a;
      }
      float m  = fmaxf(o[0], fmaxf(o[1], o[2]));
      float e0 = expf(o[0] - m), e1 = expf(o[1] - m), e2 = expf(o[2] - m);
      float inv = 1.0f / (e0 + e1 + e2);
      Out[(size_t)(bm + tid) * 3 + 0] = e0 * inv;
      Out[(size_t)(bm + tid) * 3 + 1] = e1 * inv;
      Out[(size_t)(bm + tid) * 3 + 2] = e2 * inv;
    }
  }
}

// ===========================================================================
// shift = sigmoid(latent @ W_shift + b) * 0.01     (one block per row)
// ===========================================================================
__global__ __launch_bounds__(256)
void k_shift(const float* __restrict__ X, const float* __restrict__ W,
             const float* __restrict__ b, float* __restrict__ out) {
  __shared__ float red[256];
  const int r = blockIdx.x;
  red[threadIdx.x] = X[(size_t)r * kL + threadIdx.x] * W[threadIdx.x];
  __syncthreads();
  for (int o = 128; o > 0; o >>= 1) {
    if (threadIdx.x < o) red[threadIdx.x] += red[threadIdx.x + o];
    __syncthreads();
  }
  if (threadIdx.x == 0)
    out[r] = 0.01f / (1.0f + expf(-(red[0] + b[0])));
}

// ===========================================================================
// Row-wise unit_norm (x / (||x|| + 1e-8)) in place.
// ===========================================================================
__global__ __launch_bounds__(256)
void k_unitnorm(float* __restrict__ p, int len) {
  __shared__ float red[256];
  const int r = blockIdx.x;
  float s = 0.f;
  for (int i = threadIdx.x; i < len; i += 256) {
    float v = p[(size_t)r * len + i];
    s += v * v;
  }
  red[threadIdx.x] = s;
  __syncthreads();
  for (int o = 128; o > 0; o >>= 1) {
    if (threadIdx.x < o) red[threadIdx.x] += red[threadIdx.x + o];
    __syncthreads();
  }
  float inv = 1.0f / (sqrtf(red[0]) + 1e-8f);
  for (int i = threadIdx.x; i < len; i += 256)
    p[(size_t)r * len + i] *= inv;
}

// ===========================================================================
// In-place LDS radix-2 complex FFT (block cooperative), n = power of two.
// ===========================================================================
__device__ void block_bitrev(float2* s, int n, int logn) {
  for (int i = threadIdx.x; i < n; i += blockDim.x) {
    int j = __brev((unsigned)i) >> (32 - logn);
    if (j > i) { float2 t = s[i]; s[i] = s[j]; s[j] = t; }
  }
  __syncthreads();
}

__device__ void block_fft(float2* s, int n, int logn, float sign) {
  block_bitrev(s, n, logn);
  for (int len = 2; len <= n; len <<= 1) {
    const int   half = len >> 1;
    const float ang  = sign * 6.28318530717958647692f / (float)len;
    for (int i = threadIdx.x; i < (n >> 1); i += blockDim.x) {
      int blk = i / half, pos = i - blk * half;
      int j   = blk * len + pos;
      float ss, cc;
      __sincosf(ang * (float)pos, &ss, &cc);
      float2 a = s[j], b = s[j + half];
      float2 t = make_float2(cc * b.x - ss * b.y, cc * b.y + ss * b.x);
      s[j]        = make_float2(a.x + t.x, a.y + t.y);
      s[j + half] = make_float2(a.x - t.x, a.y - t.y);
    }
    __syncthreads();
  }
}

// real-FFT pack/unpack (length-N real transform via length-M=N/2 complex FFT)
__device__ __forceinline__ float2 rfft_unpack(float2 Zk, float2 ZMk, int k, int N) {
  float2 Xe = make_float2(0.5f * (Zk.x + ZMk.x), 0.5f * (Zk.y - ZMk.y));
  float2 Xd = make_float2(0.5f * (Zk.x - ZMk.x), 0.5f * (Zk.y + ZMk.y));
  float2 Xo = make_float2(Xd.y, -Xd.x);                 // -i * Xd
  float ss, cc;
  __sincosf(-6.28318530717958647692f * (float)k / (float)N, &ss, &cc);
  return make_float2(Xe.x + cc * Xo.x - ss * Xo.y,
                     Xe.y + cc * Xo.y + ss * Xo.x);     // Xe + w_k * Xo
}

__device__ __forceinline__ float2 rfft_pack(float2 Xk, float2 XMk, int k, int N) {
  float2 Xe = make_float2(0.5f * (Xk.x + XMk.x), 0.5f * (Xk.y - XMk.y));
  float2 Xd = make_float2(0.5f * (Xk.x - XMk.x), 0.5f * (Xk.y + XMk.y));
  float ss, cc;
  __sincosf(6.28318530717958647692f * (float)k / (float)N, &ss, &cc);
  float2 Xo = make_float2(cc * Xd.x - ss * Xd.y, cc * Xd.y + ss * Xd.x); // conj(w_k)*Xd
  return make_float2(Xe.x - Xo.y, Xe.y + Xo.x);         // Xe + i*Xo
}

// ===========================================================================
// fft_frequency_recompose: one block per bank row (64 rows).
// LDS: spec (16385) + work (<=16384) complex = ~256KB.
// ===========================================================================
__global__ __launch_bounds__(256)
void k_recompose(const float* __restrict__ b0, const float* __restrict__ b1,
                 const float* __restrict__ b2, const float* __restrict__ b3,
                 const float* __restrict__ b4, const float* __restrict__ b5,
                 const float* __restrict__ b6, float* __restrict__ out) {
  extern __shared__ float2 sm[];
  float2* spec = sm;                 // 16385 coeffs of the 32768-pt rFFT
  float2* work = sm + 16400;         // up to 16384 packed complex
  const int r = blockIdx.x;
  const float* bands[7] = {b0, b1, b2, b3, b4, b5, b6};
  const int    sizes[7] = {512, 1024, 2048, 4096, 8192, 16384, 32768};

  for (int k = threadIdx.x; k <= 16384; k += 256)
    spec[k] = make_float2(0.f, 0.f);
  __syncthreads();

  for (int b = 0; b < 7; ++b) {
    const int s  = sizes[b];
    const int Mh = s >> 1;
    const int lg = 31 - __clz(Mh);
    const float* src = bands[b] + (size_t)r * s;
    for (int i = threadIdx.x * 4; i < s; i += 1024)     // async row -> LDS
      async_copy16(src + i, (float*)work + i);
    async_wait<0>();
    __syncthreads();
    block_fft(work, Mh, lg, -1.f);
    const int   nc  = Mh + 1;
    const float scl = rsqrtf((float)s);                 // rfft(norm='ortho')
    for (int k = threadIdx.x; k <= Mh; k += 256) {
      float2 X = rfft_unpack(work[k & (Mh - 1)], work[(Mh - k) & (Mh - 1)], k, s);
      if (b > 0 && k < (nc >> 1)) X = make_float2(0.f, 0.f);  // keep top half
      spec[k] = make_float2(spec[k].x + X.x * scl, spec[k].y + X.y * scl);
    }
    __syncthreads();
  }

  // irfft(n=32768, norm='ortho'): pack 16385 coeffs -> 16384 complex, inverse
  for (int k = threadIdx.x; k < 16384; k += 256)
    work[k] = rfft_pack(spec[k], spec[16384 - k], k, kRES);
  __syncthreads();
  block_fft(work, 16384, 14, +1.f);
  const float scl = rsqrtf((float)kRES) * (1.0f / 16384.f);  // ortho inverse
  for (int k = threadIdx.x; k < 16384; k += 256) {
    out[(size_t)r * kRES + 2 * k]     = work[k].x * scl;
    out[(size_t)r * kRES + 2 * k + 1] = work[k].y * scl;
  }
}

// ===========================================================================
// Forward 65536-pt rFFT of a zero-padded 32768 row -> spectrum[32769] complex.
// Async copy row (128KB contiguous) into LDS; zero-pad upper half via DS.
// ===========================================================================
__global__ __launch_bounds__(256)
void k_rfft64k(const float* __restrict__ src, float2* __restrict__ spec) {
  extern __shared__ float2 z[];
  const int r = blockIdx.x;
  {
    const float* g  = src + (size_t)r * kRES;
    float*       zl = (float*)z;
    for (int i = threadIdx.x * 4; i < kRES; i += 1024)
      async_copy16(g + i, zl + i);
    for (int k = kRES / 2 + threadIdx.x; k < kMF; k += 256)
      z[k] = make_float2(0.f, 0.f);                    // zero pad to 65536
    async_wait<0>();
  }
  __syncthreads();
  block_fft(z, kMF, 15, -1.f);
  for (int k = threadIdx.x; k <= kMF; k += 256)
    spec[(size_t)r * (kMF + 1) + k] =
        rfft_unpack(z[k & (kMF - 1)], z[(kMF - k) & (kMF - 1)], k, kNF);
}

// ===========================================================================
// conv2 = irfft(specA * specB)[:32768]   (one block per row, LDS-resident)
// ===========================================================================
__global__ __launch_bounds__(256)
void k_specmul_irfft(const float2* __restrict__ specA,
                     const float2* __restrict__ specB,
                     float* __restrict__ out) {
  extern __shared__ float2 z[];
  const int r = blockIdx.x;
  const float2* A = specA + (size_t)r * (kMF + 1);
  const float2* B = specB + (size_t)r * (kMF + 1);
  for (int k = threadIdx.x; k < kMF; k += 256) {
    float2 Xk  = cmul(A[k], B[k]);
    float2 XMk = cmul(A[kMF - k], B[kMF - k]);
    z[k] = rfft_pack(Xk, XMk, k, kNF);
  }
  __syncthreads();
  block_fft(z, kMF, 15, +1.f);
  const float scl = 1.0f / (float)kMF;                  // inverse norm
  for (int k = threadIdx.x; k < kRES / 2; k += 256)     // keep first 32768
    z[k] = make_float2(z[k].x * scl, z[k].y * scl);
  __syncthreads();
  {
    float* g  = out + (size_t)r * kRES;
    float* zl = (float*)z;
    for (int i = threadIdx.x * 4; i < kRES; i += 1024)
      async_store16(g + i, zl + i);
    async_wait<0>();
  }
}

// ===========================================================================
// final0 = mx0*imp_pad + mx1*instr + mx2*conv2   (elementwise; in-place OK)
// ===========================================================================
__global__ __launch_bounds__(256)
void k_mix(const float* __restrict__ imp, const float* __restrict__ instr,
           const float* __restrict__ conv2c, const float* __restrict__ mx,
           float* __restrict__ out) {
  size_t i = (size_t)blockIdx.x * blockDim.x + threadIdx.x;
  if (i >= (size_t)kBE * kRES) return;
  int row = (int)(i >> 15);
  int s   = (int)(i & (kRES - 1));
  float m0 = mx[row * 3], m1 = mx[row * 3 + 1], m2 = mx[row * 3 + 2];
  float ip = (s < kIMP) ? imp[(size_t)row * kIMP + s] : 0.f;
  out[i] = m0 * ip + m1 * instr[i] + m2 * conv2c[i];
}

// ===========================================================================
// fft_shift: fwd 65536-pt rFFT of zero-padded row, per-bin phase
// e^{-i*(2pi k/32769)*shift_samples}, inverse, truncate.  One LDS pass/row.
// ===========================================================================
__global__ __launch_bounds__(256)
void k_fftshift(const float* __restrict__ src, const float* __restrict__ shiftv,
                float* __restrict__ dst) {
  extern __shared__ float2 z[];
  const int r = blockIdx.x;
  {
    const float* g  = src + (size_t)r * kRES;
    float*       zl = (float*)z;
    for (int i = threadIdx.x * 4; i < kRES; i += 1024)
      async_copy16(g + i, zl + i);
    for (int k = kRES / 2 + threadIdx.x; k < kMF; k += 256)
      z[k] = make_float2(0.f, 0.f);                    // zero pad to 65536
    async_wait<0>();
  }
  __syncthreads();
  block_fft(z, kMF, 15, -1.f);

  const float sh  = shiftv[r] * ((float)kRES * 0.5f);   // shift * n * 0.5
  const float wph = 6.28318530717958647692f / 32769.0f; // phase step (nc)
  // pairwise unpack -> rotate -> repack, in place (pair (k,M-k) per thread)
  for (int k = threadIdx.x; k <= kMF / 2; k += 256) {
    int k2 = kMF - k;
    float2 Zk  = z[k & (kMF - 1)];
    float2 Zk2 = z[k2 & (kMF - 1)];
    float2 Xk  = rfft_unpack(Zk, Zk2, k, kNF);
    float2 Xk2 = rfft_unpack(Zk2, Zk, k2, kNF);
    float ss, cc;
    __sincosf(-wph * (float)k * sh, &ss, &cc);
    Xk = cmul(Xk, make_float2(cc, ss));
    __sincosf(-wph * (float)k2 * sh, &ss, &cc);
    Xk2 = cmul(Xk2, make_float2(cc, ss));
    z[k] = rfft_pack(Xk, Xk2, k, kNF);
    if (k > 0 && k < kMF / 2)
      z[k2] = rfft_pack(Xk2, Xk, k2, kNF);
  }
  __syncthreads();
  block_fft(z, kMF, 15, +1.f);
  const float scl = 1.0f / (float)kMF;
  for (int k = threadIdx.x; k < kRES / 2; k += 256)
    z[k] = make_float2(z[k].x * scl, z[k].y * scl);
  __syncthreads();
  {
    float* g  = dst + (size_t)r * kRES;
    float* zl = (float*)z;
    for (int i = threadIdx.x * 4; i < kRES; i += 1024)
      async_store16(g + i, zl + i);
    async_wait<0>();
  }
}

// ===========================================================================
// Orchestration
// ===========================================================================
extern "C" void kernel_launch(void* const* d_in, const int* in_sizes, int n_in,
                              void* d_out, int out_size, void* d_ws, size_t ws_size,
                              hipStream_t stream) {
  (void)in_sizes; (void)n_in; (void)out_size; (void)ws_size;

  const float* latent  = (const float*)d_in[0];
  const float* band[7] = {(const float*)d_in[1], (const float*)d_in[2],
                          (const float*)d_in[3], (const float*)d_in[4],
                          (const float*)d_in[5], (const float*)d_in[6],
                          (const float*)d_in[7]};
  const float* W_imp   = (const float*)d_in[8];
  const float* W_instr = (const float*)d_in[9];
  const float* selP[8] = {(const float*)d_in[10], (const float*)d_in[11],
                          (const float*)d_in[12], (const float*)d_in[13],
                          (const float*)d_in[14], (const float*)d_in[15],
                          (const float*)d_in[16], (const float*)d_in[17]};
  const float* mixP[8] = {(const float*)d_in[18], (const float*)d_in[19],
                          (const float*)d_in[20], (const float*)d_in[21],
                          (const float*)d_in[22], (const float*)d_in[23],
                          (const float*)d_in[24], (const float*)d_in[25]};
  const float* W_shift = (const float*)d_in[26];
  const float* b_shift = (const float*)d_in[27];

  float* ws = (float*)d_ws;
  size_t off = 0;
  float*  sel      = ws + off; off += (size_t)kBE * 64;
  float*  mx       = ws + off; off += (size_t)kBE * 4;
  float*  shiftv   = ws + off; off += (size_t)kBE;
  off = (off + 3) & ~(size_t)3;
  float*  imp      = ws + off; off += (size_t)kBE * kIMP;
  float*  instr    = ws + off; off += (size_t)kBE * kRES;
  float*  res2full = ws + off; off += (size_t)kNB * kRES;
  float*  conv2    = ws + off; off += (size_t)kBE * kRES;
  float2* specA    = (float2*)(ws + off); off += (size_t)kBE * (kMF + 1) * 2;
  float2* specB    = (float2*)(ws + off); off += (size_t)kBE * (kMF + 1) * 2;
  // bf16 operand pool
  unsigned short* wsh = (unsigned short*)(ws + off);
  size_t hoff = 0;
  unsigned short* latA    = wsh + hoff; hoff += (size_t)kBE * kL;     // [512,256]
  unsigned short* selB    = wsh + hoff; hoff += (size_t)kBE * 64;     // [512,64]
  unsigned short* WimpT   = wsh + hoff; hoff += (size_t)kIMP * kL;    // [8192,256]
  unsigned short* WinstrT = wsh + hoff; hoff += (size_t)kRES * kL;    // [32768,256]
  unsigned short* res2fT  = wsh + hoff; hoff += (size_t)kRES * kNB;   // [32768,64]

  float* final_out = (float*)d_out;                       // [512, 32768]
  float* res2_out  = (float*)d_out + (size_t)kBE * kRES;  // [512, 32768]

  const size_t ldsFFT = (size_t)kMF * sizeof(float2);             // 256 KB
  const size_t ldsRC  = (size_t)(16400 + 16384) * sizeof(float2); // ~256 KB

  // 0) one-time bf16 operand preparation (halves GEMM weight bandwidth)
  k_cvt<<<(unsigned)((kBE * kL + 255) / 256), 256, 0, stream>>>(
      latent, latA, (size_t)kBE * kL);
  k_cvt_t<<<dim3(kIMP / 32, kL / 32), 256, 0, stream>>>(W_imp, WimpT, kL, kIMP);
  k_cvt_t<<<dim3(kRES / 32, kL / 32), 256, 0, stream>>>(W_instr, WinstrT, kL, kRES);

  // 1) control nets on latent (WMMA)
  k_stack<<<kBE / 16, 256, 0, stream>>>(latent, selP[0], selP[1], selP[2],
      selP[3], selP[4], selP[5], selP[6], selP[7], sel, 64, 0);
  k_stack<<<kBE / 16, 256, 0, stream>>>(latent, mixP[0], mixP[1], mixP[2],
      mixP[3], mixP[4], mixP[5], mixP[6], mixP[7], mx, 3, 1);
  k_shift<<<kBE, 256, 0, stream>>>(latent, W_shift, b_shift, shiftv);
  k_cvt<<<(unsigned)((kBE * 64 + 255) / 256), 256, 0, stream>>>(
      sel, selB, (size_t)kBE * 64);

  // 2) fat projections (WMMA, K=256, async double-buffered staging)
  k_wmma_gemm_bf16<<<dim3(kIMP / GT_N, kBE / GT_M), 256, 0, stream>>>(
      latA, WimpT, imp, kBE, kIMP, kL);
  k_wmma_gemm_bf16<<<dim3(kRES / GT_N, kBE / GT_M), 256, 0, stream>>>(
      latA, WinstrT, instr, kBE, kRES, kL);
  k_unitnorm<<<kBE, 256, 0, stream>>>(imp, kIMP);

  // 3) resonance bank recompose (LDS-resident FFTs)
  k_recompose<<<kNB, 256, ldsRC, stream>>>(band[0], band[1], band[2], band[3],
                                           band[4], band[5], band[6], res2full);
  k_cvt_t<<<dim3(kRES / 32, kNB / 32), 256, 0, stream>>>(
      res2full, res2fT, kNB, kRES);

  // 4) per-event resonance selection: res2 = sel @ res2full (WMMA, K=64)
  k_wmma_gemm_bf16<<<dim3(kRES / GT_N, kBE / GT_M), 256, 0, stream>>>(
      selB, res2fT, res2_out, kBE, kRES, kNB);
  k_unitnorm<<<kBE, 256, 0, stream>>>(res2_out, kRES);

  // 5) fft_convolve(instr, res2): fwd spectra, product, inverse
  k_rfft64k<<<kBE, 256, ldsFFT, stream>>>(instr, specA);
  k_rfft64k<<<kBE, 256, ldsFFT, stream>>>(res2_out, specB);
  k_specmul_irfft<<<kBE, 256, ldsFFT, stream>>>(specA, specB, conv2);

  // 6) mix [imp_pad, instr, conv2] by softmax weights (in place on conv2)
  {
    size_t total = (size_t)kBE * kRES;
    k_mix<<<(unsigned)((total + 255) / 256), 256, 0, stream>>>(
        imp, instr, conv2, mx, conv2);
  }

  // 7) spectral phase shift -> final output
  k_fftshift<<<kBE, 256, ldsFFT, stream>>>(conv2, shiftv, final_out);
}
